// TEM_T_80934363726607
// MI455X (gfx1250) — compile-verified, matrix operand order checked by reading, not run on hardware
//
#include <hip/hip_runtime.h>
#include <hip/hip_bf16.h>

// Problem constants
constexpr int kB  = 4;
constexpr int kL  = 2048;
constexpr int kD  = 256;
constexpr int kH  = 4;
constexpr int kDH = 64;
constexpr int kV  = 4096;
constexpr int kA  = 4;
constexpr int kDF = 1024;   // 4*D
constexpr float kEPS = 1e-5f;

typedef __attribute__((ext_vector_type(16))) __bf16 v16bf;
typedef __attribute__((ext_vector_type(8)))  float  v8f;

// ---------- helpers ----------
__device__ __forceinline__ __bf16 f2bf(float f) {
  union { float f; unsigned u; } a; a.f = f;
  unsigned r = a.u + 0x7FFFu + ((a.u >> 16) & 1u);   // RNE
  unsigned short h = (unsigned short)(r >> 16);
  return __builtin_bit_cast(__bf16, h);
}

// async copy of 16 bytes global -> LDS (GLOBAL_LOAD_ASYNC_TO_LDS_B128, GVS mode)
__device__ __forceinline__ void async_cp16(const void* lds_dst, unsigned voff_bytes,
                                           unsigned long long gbase) {
  unsigned laddr = (unsigned)(unsigned long long)(uintptr_t)lds_dst; // low 32b = LDS addr
  asm volatile("global_load_async_to_lds_b128 %0, %1, %2"
               :: "v"(laddr), "v"(voff_bytes), "s"(gbase) : "memory");
}
__device__ __forceinline__ void wait_async0() {
  asm volatile("s_wait_asynccnt 0x0" ::: "memory");
}
__device__ __forceinline__ void wait_async4() {
  asm volatile("s_wait_asynccnt 0x4" ::: "memory");
}

// Load one WMMA 16-bit fragment from an LDS row (row base must be 16B aligned).
// ISA 7.12.2: lane<16 holds K in {h*8..h*8+7} U {h*8+16..h*8+23}, h=lane>>4.
__device__ __forceinline__ v16bf load_frag(const __bf16* rowbase, int half) {
  union { v16bf v; uint4 q[2]; } u;
  u.q[0] = *(const uint4*)((const char*)rowbase + half * 16);
  u.q[1] = *(const uint4*)((const char*)rowbase + half * 16 + 32);
  return u.v;
}

// ---------- weight convert + transpose: W f32[K,N] -> Wt bf16[N,K] ----------
__global__ __launch_bounds__(256) void transpose_w(const float* __restrict__ W,
                                                   __bf16* __restrict__ Wt,
                                                   int K, int N) {
  __shared__ float tile[64][65];
  const int k0 = blockIdx.y * 64, n0 = blockIdx.x * 64;
  const int tx = threadIdx.x & 63, ty = threadIdx.x >> 6;
  #pragma unroll
  for (int r = ty; r < 64; r += 4)
    tile[r][tx] = W[(size_t)(k0 + r) * N + n0 + tx];
  __syncthreads();
  #pragma unroll
  for (int r = ty; r < 64; r += 4)
    Wt[(size_t)(n0 + r) * K + k0 + tx] = f2bf(tile[tx][r]);
}

// ---------- token embedding gather (f32 + bf16 copies) ----------
__global__ __launch_bounds__(256) void gather_embed(const int* __restrict__ tokens,
                                                    const float* __restrict__ emb,
                                                    float* __restrict__ X,
                                                    __bf16* __restrict__ Xb) {
  int i = blockIdx.x * 256 + threadIdx.x;     // total = B*L*D
  int row = i >> 8, col = i & 255;
  int tok = tokens[row];
  float v = emb[(size_t)tok * kD + col];
  X[i] = v;
  Xb[i] = f2bf(v);
}

// ---------- sequential recurrent scan (one block per batch element) ----------
// Tokens with tok<A are ~0.1% of positions: process 32 steps per wave-ballot;
// all-inactive groups emit rows with zero barriers and zero scalar loads.
__global__ __launch_bounds__(256) void scan_rnn(const int* __restrict__ tokens,
                                                const float* __restrict__ e_init,
                                                const float* __restrict__ W_a,
                                                const float* __restrict__ g,
                                                const float* __restrict__ bta,
                                                float* __restrict__ E) {
  __shared__ float es[kD];
  __shared__ float red[8][2];
  const int tid = threadIdx.x, b = blockIdx.x;
  const int wid = tid >> 5, lane = tid & 31;
  float e_t = e_init[tid];
  es[tid] = e_t;
  __syncthreads();
  float* Eb = E + (size_t)b * kL * kD;

  for (int t0 = 0; t0 < kL; t0 += 32) {
    // every wave loads the same 32 tokens -> identical ballot in all waves
    const int tokv = tokens[b * kL + t0 + lane];
    const unsigned long long mask = __ballot(tokv < kA);
    if (mask == 0ull) {
      // 32 inactive steps: rows are all the current state
      #pragma unroll
      for (int j = 0; j < 32; ++j)
        Eb[(size_t)(t0 + j) * kD + tid] = e_t;
    } else {
      for (int j = 0; j < 32; ++j) {
        const int tok = __shfl(tokv, j, 32);   // uniform across block
        if (tok < kA) {
          const float* w = W_a + (size_t)tok * kD * kD + tid;
          float acc = 0.f;
          #pragma unroll 4
          for (int k = 0; k < kD; ++k) acc = fmaf(es[k], w[(size_t)k * kD], acc);
          const float r = fmaxf(acc, 0.f);
          float s = r, s2 = r * r;
          #pragma unroll
          for (int off = 16; off >= 1; off >>= 1) {
            s  += __shfl_xor(s,  off, 32);
            s2 += __shfl_xor(s2, off, 32);
          }
          if (lane == 0) { red[wid][0] = s; red[wid][1] = s2; }
          __syncthreads();
          float ts = 0.f, ts2 = 0.f;
          #pragma unroll
          for (int wv = 0; wv < 8; ++wv) { ts += red[wv][0]; ts2 += red[wv][1]; }
          const float mean = ts * (1.f / kD);
          const float var  = ts2 * (1.f / kD) - mean * mean;
          e_t = (r - mean) * rsqrtf(var + kEPS) * g[tid] + bta[tid];
          __syncthreads();      // red consumed; es reads done
          es[tid] = e_t;
          __syncthreads();      // es update visible before next active step
        }
        Eb[(size_t)(t0 + j) * kD + tid] = e_t;
      }
    }
  }
}

// ---------- row LayerNorm over D=256 (wave per row); optional f32/bf16 outs ----------
__global__ __launch_bounds__(256) void ln_rows(const float* __restrict__ Xin,
                                               const float* __restrict__ g,
                                               const float* __restrict__ bta,
                                               float* __restrict__ outF,
                                               __bf16* __restrict__ outH) {
  const int tid = threadIdx.x, wid = tid >> 5, lane = tid & 31;
  const size_t row = (size_t)blockIdx.x * 8 + wid;
  const float* xr = Xin + row * kD + lane * 8;
  float x[8];
  #pragma unroll
  for (int j = 0; j < 8; ++j) x[j] = xr[j];
  float s = 0.f, s2 = 0.f;
  #pragma unroll
  for (int j = 0; j < 8; ++j) { s += x[j]; s2 += x[j] * x[j]; }
  #pragma unroll
  for (int off = 16; off >= 1; off >>= 1) {
    s  += __shfl_xor(s,  off, 32);
    s2 += __shfl_xor(s2, off, 32);
  }
  const float mean = s * (1.f / kD);
  const float var  = s2 * (1.f / kD) - mean * mean;
  const float inv  = rsqrtf(var + kEPS);
  #pragma unroll
  for (int j = 0; j < 8; ++j) {
    const int c = lane * 8 + j;
    const float y = (x[j] - mean) * inv * g[c] + bta[c];
    if (outF) outF[row * kD + c] = y;
    if (outH) outH[row * kD + c] = f2bf(y);
  }
}

// ---------- WMMA GEMM: out = epi(A[M,K]bf16 @ Wt[N,K]^T + bias + R) ----------
// Block tile 128x128, 8 waves; each wave one 16x128 strip (8 accumulators).
// Double-buffered async LDS staging. EPI: 0 = identity, 1 = exact-erf GELU.
template <int EPI>
__global__ __launch_bounds__(256) void gemm_bf(const __bf16* __restrict__ A,
                                               const __bf16* __restrict__ Wt,
                                               const float* __restrict__ bias,
                                               const float* __restrict__ R,
                                               float* __restrict__ outF,
                                               __bf16* __restrict__ outH,
                                               int M, int N, int K) {
  __shared__ __align__(16) __bf16 At[2][128 * 32];
  __shared__ __align__(16) __bf16 Bt[2][128 * 32];
  const int tid = threadIdx.x;
  const int wid = tid >> 5, lane = tid & 31;
  const int half = lane >> 4, r16 = lane & 15;
  const int m0 = blockIdx.y * 128, n0 = blockIdx.x * 128;
  const unsigned long long baseA = (unsigned long long)(uintptr_t)A;
  const unsigned long long baseW = (unsigned long long)(uintptr_t)Wt;

  auto issue = [&](int k0, int buf) {
    #pragma unroll
    for (int i = 0; i < 2; ++i) {           // A tile: 128x32 = 512 chunks of 16B
      const int c = tid * 2 + i;
      const int r = c >> 2, cc = (c & 3) * 8;
      const unsigned voff = ((unsigned)(m0 + r) * (unsigned)K + (unsigned)(k0 + cc)) * 2u;
      async_cp16(&At[buf][r * 32 + cc], voff, baseA);
    }
    #pragma unroll
    for (int i = 0; i < 2; ++i) {           // B tile: 128x32 = 512 chunks
      const int c = tid * 2 + i;
      const int n = c >> 2, cc = (c & 3) * 8;
      const unsigned voff = ((unsigned)(n0 + n) * (unsigned)K + (unsigned)(k0 + cc)) * 2u;
      async_cp16(&Bt[buf][n * 32 + cc], voff, baseW);
    }
  };

  v8f acc[8] = {};
  const int nk = K >> 5;
  issue(0, 0);
  for (int it = 0; it < nk; ++it) {
    const int cur = it & 1;
    const bool more = (it + 1) < nk;
    if (more) { issue((it + 1) * 32, cur ^ 1); wait_async4(); }
    else      { wait_async0(); }
    __syncthreads();                         // all waves' tile-cur copies landed
    const v16bf a = load_frag(&At[cur][(wid * 16 + r16) * 32], half);
    #pragma unroll
    for (int j = 0; j < 8; ++j) {
      const v16bf b = load_frag(&Bt[cur][(j * 16 + r16) * 32], half);
      acc[j] = __builtin_amdgcn_wmma_f32_16x16x32_bf16(false, a, false, b,
                                                       (short)0, acc[j], false, false);
    }
    __syncthreads();                         // done reading buf cur
  }

  const int row_base = m0 + wid * 16 + half * 8;
  #pragma unroll
  for (int j = 0; j < 8; ++j) {
    const int col = n0 + j * 16 + r16;
    const float bval = bias ? bias[col] : 0.f;
    #pragma unroll
    for (int v = 0; v < 8; ++v) {
      const size_t row = row_base + v;
      float x = acc[j][v] + bval;
      if (R) x += R[row * N + col];
      if (EPI == 1) x = 0.5f * x * (1.0f + erff(x * 0.70710678118654752f));
      if (outF) outF[row * N + col] = x;
      if (outH) outH[row * N + col] = f2bf(x);
    }
  }
}

// ---------- flash-style causal attention, Q==K (bf16 in), per-head ----------
// grid: (L/128, H, B); block 256 (8 waves); wave handles 16 q rows.
__global__ __launch_bounds__(256) void attn_flash(const __bf16* __restrict__ Q,
                                                  const __bf16* __restrict__ Vv,
                                                  __bf16* __restrict__ Y) {
  __shared__ __align__(16) __bf16 Qs[128 * 64];
  __shared__ __align__(16) __bf16 Ks[32 * 64];
  __shared__ __align__(16) __bf16 VTs[64 * 32];
  __shared__ __align__(16) __bf16 Ps[8][16 * 32];

  const int tid = threadIdx.x, wid = tid >> 5, lane = tid & 31;
  const int half = lane >> 4, r16 = lane & 15;
  const int b = blockIdx.z, h = blockIdx.y, q0 = blockIdx.x * 128;
  const __bf16* Qh = Q  + ((size_t)b * kL) * kD + h * kDH;
  const __bf16* Vh = Vv + ((size_t)b * kL) * kD + h * kDH;
  const unsigned long long baseQ = (unsigned long long)(uintptr_t)Qh;

  { // stage Q tile 128x64 async: 1024 chunks of 16B
    #pragma unroll
    for (int i = 0; i < 4; ++i) {
      const int c = tid + i * 256;
      const int r = c >> 3, cc = (c & 7) * 8;
      const unsigned voff = ((unsigned)(q0 + r) * kD + (unsigned)cc) * 2u;
      async_cp16(&Qs[r * 64 + cc], voff, baseQ);
    }
  }

  const int wq0 = q0 + wid * 16;
  v8f accO[4] = {};
  float mrow[8], lrow[8];
  #pragma unroll
  for (int v = 0; v < 8; ++v) { mrow[v] = -1e30f; lrow[v] = 0.f; }

  const int nkt = (q0 + 128) >> 5;
  for (int kt = 0; kt < nkt; ++kt) {
    const int kb = kt * 32;
    const int key = tid >> 3, cc = (tid & 7) * 8;
    { // K tile async (K == Q)
      const unsigned voff = ((unsigned)(kb + key) * kD + (unsigned)cc) * 2u;
      async_cp16(&Ks[key * 64 + cc], voff, baseQ);
    }
    { // V tile: coalesced b128 load, transposed scalar scatter into VTs
      const uint4 raw = *(const uint4*)(Vh + (size_t)(kb + key) * kD + cc);
      const unsigned short* e = (const unsigned short*)&raw;
      #pragma unroll
      for (int j = 0; j < 8; ++j)
        VTs[(cc + j) * 32 + key] = __builtin_bit_cast(__bf16, e[j]);
    }
    wait_async0();
    __syncthreads();

    if (kb <= wq0 + 15) {
      v8f accS[2] = {};
      #pragma unroll
      for (int kk = 0; kk < 2; ++kk) {
        const v16bf aq = load_frag(&Qs[(wid * 16 + r16) * 64 + kk * 32], half);
        #pragma unroll
        for (int j = 0; j < 2; ++j) {
          const v16bf bk = load_frag(&Ks[(j * 16 + r16) * 64 + kk * 32], half);
          accS[j] = __builtin_amdgcn_wmma_f32_16x16x32_bf16(false, aq, false, bk,
                                                            (short)0, accS[j], false, false);
        }
      }
      float scale[8];
      #pragma unroll
      for (int v = 0; v < 8; ++v) {
        const int qrow = wq0 + half * 8 + v;
        float s0 = accS[0][v] * 0.125f;            // 1/sqrt(64)
        float s1 = accS[1][v] * 0.125f;
        if (kb + r16 > qrow)      s0 = -1e30f;     // causal mask
        if (kb + 16 + r16 > qrow) s1 = -1e30f;
        float mx = fmaxf(s0, s1);
        #pragma unroll
        for (int off = 8; off >= 1; off >>= 1) mx = fmaxf(mx, __shfl_xor(mx, off, 32));
        const float mnew = fmaxf(mrow[v], mx);
        const float p0 = __expf(s0 - mnew);
        const float p1 = __expf(s1 - mnew);
        accS[0][v] = p0; accS[1][v] = p1;
        float ps = p0 + p1;
        #pragma unroll
        for (int off = 8; off >= 1; off >>= 1) ps += __shfl_xor(ps, off, 32);
        scale[v] = __expf(mrow[v] - mnew);
        lrow[v] = lrow[v] * scale[v] + ps;
        mrow[v] = mnew;
      }
      #pragma unroll
      for (int j = 0; j < 4; ++j)
        #pragma unroll
        for (int v = 0; v < 8; ++v) accO[j][v] *= scale[v];
      __bf16* Pw = Ps[wid];
      #pragma unroll
      for (int v = 0; v < 8; ++v) {
        Pw[(half * 8 + v) * 32 + r16]      = f2bf(accS[0][v]);
        Pw[(half * 8 + v) * 32 + 16 + r16] = f2bf(accS[1][v]);
      }
      const v16bf ap = load_frag(&Pw[r16 * 32], half);
      #pragma unroll
      for (int j = 0; j < 4; ++j) {
        const v16bf bv = load_frag(&VTs[(j * 16 + r16) * 32], half);
        accO[j] = __builtin_amdgcn_wmma_f32_16x16x32_bf16(false, ap, false, bv,
                                                          (short)0, accO[j], false, false);
      }
    }
    __syncthreads();
  }

  __bf16* Yh = Y + ((size_t)b * kL) * kD + h * kDH;
  #pragma unroll
  for (int j = 0; j < 4; ++j) {
    #pragma unroll
    for (int v = 0; v < 8; ++v) {
      const int qrow = wq0 + half * 8 + v;
      Yh[(size_t)qrow * kD + j * 16 + r16] = f2bf(accO[j][v] / lrow[v]);
    }
  }
}

// ---------------------------------------------------------------------------
extern "C" void kernel_launch(void* const* d_in, const int* in_sizes, int n_in,
                              void* d_out, int out_size, void* d_ws, size_t ws_size,
                              hipStream_t stream) {
  (void)in_sizes; (void)n_in; (void)out_size; (void)ws_size;
  const int*   tokens   = (const int*)  d_in[0];
  const float* stim_emb = (const float*)d_in[1];
  const float* e_init   = (const float*)d_in[2];
  const float* W_a      = (const float*)d_in[3];
  const float* e_rnn_g  = (const float*)d_in[4];
  const float* e_rnn_b  = (const float*)d_in[5];
  const float* pre_g    = (const float*)d_in[6];
  const float* pre_b    = (const float*)d_in[7];
  const float* W_e      = (const float*)d_in[8];
  const float* W_x      = (const float*)d_in[9];
  const float* o_w      = (const float*)d_in[10];
  const float* o_b      = (const float*)d_in[11];
  const float* n1_g     = (const float*)d_in[12];
  const float* n1_b     = (const float*)d_in[13];
  const float* n2_g     = (const float*)d_in[14];
  const float* n2_b     = (const float*)d_in[15];
  const float* f_w1     = (const float*)d_in[16];
  const float* f_b1     = (const float*)d_in[17];
  const float* f_w2     = (const float*)d_in[18];
  const float* f_b2     = (const float*)d_in[19];
  const float* on_g     = (const float*)d_in[20];
  const float* on_b     = (const float*)d_in[21];
  const float* op_w     = (const float*)d_in[22];
  const float* op_b     = (const float*)d_in[23];

  char* ws = (char*)d_ws;
  const size_t MB = 1u << 20, KBB = 1u << 10;
  float*  E   = (float*) (ws + 0 * MB);     // 8MB scan out
  float*  X   = (float*) (ws + 8 * MB);     // 8MB embeddings f32 (residual)
  __bf16* Xb  = (__bf16*)(ws + 16 * MB);    // 4MB embeddings bf16
  __bf16* Enb = (__bf16*)(ws + 20 * MB);    // 4MB ln(E) bf16
  __bf16* Qb  = (__bf16*)(ws + 24 * MB);    // 4MB Q(=K) bf16
  __bf16* Vb  = (__bf16*)(ws + 28 * MB);    // 4MB V bf16
  __bf16* Yb  = (__bf16*)(ws + 32 * MB);    // 4MB attn out bf16
  float*  s1  = (float*) (ws + 0 * MB);     // reuse E
  float*  y1  = (float*) (ws + 8 * MB);     // reuse X (dead after s1 gemm)
  __bf16* y1b = (__bf16*)(ws + 16 * MB);    // reuse Xb
  float*  s2  = (float*) (ws + 20 * MB);    // reuse Enb+Qb
  float*  y2  = (float*) (ws + 28 * MB);    // reuse Vb+Yb
  __bf16* hb  = (__bf16*)(ws + 36 * MB);    // 16MB FFN hidden bf16
  __bf16* y3b = (__bf16*)(ws + 52 * MB);    // 4MB
  __bf16* Wet = (__bf16*)(ws + 56 * MB);               // 128KB each for DxD
  __bf16* Wxt = (__bf16*)(ws + 56 * MB + 128 * KBB);
  __bf16* owt = (__bf16*)(ws + 56 * MB + 256 * KBB);
  __bf16* w1t = (__bf16*)(ws + 56 * MB + 384 * KBB);   // 512KB
  __bf16* w2t = (__bf16*)(ws + 56 * MB + 896 * KBB);   // 512KB
  __bf16* opt = (__bf16*)(ws + 56 * MB + 1408 * KBB);  // 2MB

  const int M = kB * kL;  // 8192
  dim3 blk(256);

  // 0) weight convert + transpose to bf16 [N][K]
  transpose_w<<<dim3(kD / 64, kD / 64),  blk, 0, stream>>>(W_e,  Wet, kD,  kD);
  transpose_w<<<dim3(kD / 64, kD / 64),  blk, 0, stream>>>(W_x,  Wxt, kD,  kD);
  transpose_w<<<dim3(kD / 64, kD / 64),  blk, 0, stream>>>(o_w,  owt, kD,  kD);
  transpose_w<<<dim3(kDF / 64, kD / 64), blk, 0, stream>>>(f_w1, w1t, kD,  kDF);
  transpose_w<<<dim3(kD / 64, kDF / 64), blk, 0, stream>>>(f_w2, w2t, kDF, kD);
  transpose_w<<<dim3(kV / 64, kD / 64),  blk, 0, stream>>>(op_w, opt, kD,  kV);
  // 1) X = stim_emb[tokens]
  gather_embed<<<dim3((kB * kL * kD) / 256), blk, 0, stream>>>(tokens, stim_emb, X, Xb);
  // 2) E = scan (ballot fast path)
  scan_rnn<<<dim3(kB), blk, 0, stream>>>(tokens, e_init, W_a, e_rnn_g, e_rnn_b, E);
  // 3) Enb = ln(E) (bf16)
  ln_rows<<<dim3(M / 8), blk, 0, stream>>>(E, pre_g, pre_b, nullptr, Enb);
  // 4) Q = Enb @ W_e   (K == Q)
  gemm_bf<0><<<dim3(kD / 128, M / 128), blk, 0, stream>>>(Enb, Wet, nullptr, nullptr, nullptr, Qb, M, kD, kD);
  // 5) V = Xb @ W_x
  gemm_bf<0><<<dim3(kD / 128, M / 128), blk, 0, stream>>>(Xb, Wxt, nullptr, nullptr, nullptr, Vb, M, kD, kD);
  // 6) attention -> Yb bf16
  attn_flash<<<dim3(kL / 128, kH, kB), blk, 0, stream>>>(Qb, Vb, Yb);
  // 7) s1 = Yb @ o_w + o_b + X
  gemm_bf<0><<<dim3(kD / 128, M / 128), blk, 0, stream>>>(Yb, owt, o_b, X, s1, nullptr, M, kD, kD);
  // 8) y1 = ln(s1) (f32 + bf16)
  ln_rows<<<dim3(M / 8), blk, 0, stream>>>(s1, n1_g, n1_b, y1, y1b);
  // 9) h = gelu(y1b @ w1 + b1) -> bf16
  gemm_bf<1><<<dim3(kDF / 128, M / 128), blk, 0, stream>>>(y1b, w1t, f_b1, nullptr, nullptr, hb, M, kDF, kD);
  // 10) s2 = h @ w2 + b2 + y1
  gemm_bf<0><<<dim3(kD / 128, M / 128), blk, 0, stream>>>(hb, w2t, f_b2, y1, s2, nullptr, M, kD, kDF);
  // 11) y2 = ln(s2)
  ln_rows<<<dim3(M / 8), blk, 0, stream>>>(s2, n2_g, n2_b, y2, nullptr);
  // 12) y3 = ln(y2, out_norm) (bf16)
  ln_rows<<<dim3(M / 8), blk, 0, stream>>>(y2, on_g, on_b, nullptr, y3b);
  // 13) out = y3 @ op_w + op_b
  gemm_bf<0><<<dim3(kV / 128, M / 128), blk, 0, stream>>>(y3b, opt, op_b, nullptr, (float*)d_out, nullptr, M, kV, kD);
}
